// RealSwitchMoE_16544214024857
// MI455X (gfx1250) — matrix operand order repaired and checked by reading, not compiled, not run on hardware
//
#include <hip/hip_runtime.h>
#include <hip/hip_bf16.h>

typedef __attribute__((ext_vector_type(16))) _Float16 v16h;
typedef __attribute__((ext_vector_type(8)))  float    v8f;

#define T_TOK   8192
#define D_MODEL 1024
#define D_FF    4096
#define NEXP    8
#define CAP     1024

#define BM 128
#define BN 128
#define BK 32

union FragU {
    uint4 u[2];
    v16h  h;
};

// async global->LDS copy of 32 bytes per lane as two b128 ops sharing one
// address pair (INST_OFFSET applies to both LDS and global sides on CDNA5)
__device__ __forceinline__ void async_stage32(unsigned int lds_off,
                                              const _Float16* gptr) {
    asm volatile("global_load_async_to_lds_b128 %0, %1, off\n\t"
                 "global_load_async_to_lds_b128 %0, %1, off offset:16"
                 :: "v"(lds_off), "v"((unsigned long long)(uintptr_t)gptr)
                 : "memory");
}

// ---------------------------------------------------------------------------
// 1. Transpose f32 [R][C] -> f16 [C][R], per expert (blockIdx.z)
// ---------------------------------------------------------------------------
__global__ void transpose_cvt_kernel(const float* __restrict__ in,
                                     _Float16* __restrict__ out,
                                     int R, int C) {
    __shared__ float tile[32][33];
    int e = blockIdx.z;
    const float* ine = in + (size_t)e * R * C;
    _Float16* oute = out + (size_t)e * R * C;
    int c0 = blockIdx.x * 32, r0 = blockIdx.y * 32;
    int tx = threadIdx.x, ty = threadIdx.y;
#pragma unroll
    for (int i = ty; i < 32; i += 8)
        tile[i][tx] = ine[(size_t)(r0 + i) * C + (c0 + tx)];
    __syncthreads();
#pragma unroll
    for (int i = ty; i < 32; i += 8)
        oute[(size_t)(c0 + i) * R + (r0 + tx)] = (_Float16)tile[tx][i];
}

// ---------------------------------------------------------------------------
// 2. Router: one wave (32 lanes) per token
// ---------------------------------------------------------------------------
__global__ void router_kernel(const float* __restrict__ x,
                              const float* __restrict__ Wr,
                              int* __restrict__ list_cnt,          // [E]
                              float* __restrict__ probsum,         // [E]
                              unsigned long long* __restrict__ plist) { // [E][T]
    int t = blockIdx.x * 8 + (threadIdx.x >> 5);
    int lane = threadIdx.x & 31;
    float acc[NEXP];
#pragma unroll
    for (int e = 0; e < NEXP; e++) acc[e] = 0.0f;

    for (int d = lane; d < D_MODEL; d += 32) {
        float xv = x[(size_t)t * D_MODEL + d];
        const float* wr = Wr + (size_t)d * NEXP;
#pragma unroll
        for (int e = 0; e < NEXP; e++) acc[e] += xv * wr[e];
    }
#pragma unroll
    for (int off = 16; off > 0; off >>= 1) {
#pragma unroll
        for (int e = 0; e < NEXP; e++)
            acc[e] += __shfl_xor(acc[e], off, 32);
    }
    if (lane == 0) {
        float mx = acc[0];
#pragma unroll
        for (int e = 1; e < NEXP; e++) mx = fmaxf(mx, acc[e]);
        float p[NEXP], s = 0.0f;
#pragma unroll
        for (int e = 0; e < NEXP; e++) { p[e] = __expf(acc[e] - mx); s += p[e]; }
        float inv = 1.0f / s;
        int   best = 0;
        float bp   = p[0] * inv;
#pragma unroll
        for (int e = 0; e < NEXP; e++) {
            float pe = p[e] * inv;
            atomicAdd(&probsum[e], pe);
            if (pe > bp) { bp = pe; best = e; }
        }
        int pos = atomicAdd(&list_cnt[best], 1);
        unsigned long long key =
            ((unsigned long long)__float_as_uint(bp) << 32) |
            (unsigned long long)(0xFFFFFFFFu - (unsigned)t);
        plist[(size_t)best * T_TOK + pos] = key;
    }
}

// ---------------------------------------------------------------------------
// 3. Per-expert bitonic top-k (sort 8192 keys descending, keep first 1024)
// ---------------------------------------------------------------------------
__global__ void __launch_bounds__(1024)
sort_kernel(const unsigned long long* __restrict__ plist,
            const int* __restrict__ list_cnt,
            float* __restrict__ gates, int* __restrict__ tok) {
    __shared__ unsigned long long s[T_TOK];
    int e = blockIdx.x;
    int n = list_cnt[e];
    for (int i = threadIdx.x; i < T_TOK; i += 1024)
        s[i] = (i < n) ? plist[(size_t)e * T_TOK + i] : 0ull;
    __syncthreads();
    for (int k = 2; k <= T_TOK; k <<= 1) {
        for (int j = k >> 1; j > 0; j >>= 1) {
            for (int i = threadIdx.x; i < T_TOK; i += 1024) {
                int ixj = i ^ j;
                if (ixj > i) {
                    unsigned long long a = s[i], b = s[ixj];
                    bool desc = ((i & k) == 0);
                    if (desc ? (a < b) : (a > b)) { s[i] = b; s[ixj] = a; }
                }
            }
            __syncthreads();
        }
    }
    int c = threadIdx.x;            // blockDim == CAP == 1024
    unsigned long long key = s[c];
    unsigned gb = (unsigned)(key >> 32);
    gates[e * CAP + c] = __uint_as_float(gb);
    tok[e * CAP + c]   = gb ? (int)(0xFFFFFFFFu - (unsigned)(key & 0xFFFFFFFFu)) : 0;
}

// ---------------------------------------------------------------------------
// 4. Gather selected token rows, f32 -> f16
// ---------------------------------------------------------------------------
__global__ void gather_kernel(const float* __restrict__ x,
                              const int* __restrict__ tok,
                              _Float16* __restrict__ Xe) {
    int row = blockIdx.x;                 // e*CAP + c
    int t = tok[row];
    const float4* src = (const float4*)(x + (size_t)t * D_MODEL);
    float4 v = src[threadIdx.x];
    union { _Float16 h[4]; uint2 u; } cv;
    cv.h[0] = (_Float16)v.x; cv.h[1] = (_Float16)v.y;
    cv.h[2] = (_Float16)v.z; cv.h[3] = (_Float16)v.w;
    *(uint2*)(Xe + (size_t)row * D_MODEL + threadIdx.x * 4) = cv.u;
}

// ---------------------------------------------------------------------------
// 5. Tiled WMMA GEMM with double-buffered async global->LDS staging.
//    MODE 1: H = relu(Xe*W1 + b1) -> f16
//    MODE 2: out[tok] = gate * (H*W2 + b2)
// A: [E][M][K] f16 row-major.  BT: [E][N][K] f16 row-major.
// ---------------------------------------------------------------------------
template <int MODE>
__global__ void __launch_bounds__(256)
moe_gemm_kernel(const _Float16* __restrict__ A,
                const _Float16* __restrict__ BT,
                const float* __restrict__ bias,    // [E][N]
                const float* __restrict__ gates,   // [E][CAP]
                const int* __restrict__ tok,       // [E][CAP]
                _Float16* __restrict__ Hout,       // MODE 1
                float* __restrict__ Yout,          // MODE 2
                int M, int N, int K) {
    __shared__ _Float16 As[2][BM * BK];
    __shared__ _Float16 Bs[2][BN * BK];

    int e  = blockIdx.z;
    int bm = blockIdx.y;
    int bn = blockIdx.x;
    const _Float16* Ae = A  + (size_t)e * M * K;
    const _Float16* Be = BT + (size_t)e * N * K;

    int tid  = threadIdx.x;
    int lane = tid & 31;
    int w    = tid >> 5;
    int wm   = w & 3;           // 4 waves along M -> 32 rows each
    int wn   = w >> 2;          // 2 waves along N -> 64 cols each
    int l16  = lane & 15;
    int lhi  = lane >> 4;

    v8f acc[2][4];
#pragma unroll
    for (int j = 0; j < 4; j++) {
        float b = bias[(size_t)e * N + bn * BN + wn * 64 + j * 16 + l16];
#pragma unroll
        for (int i = 0; i < 2; i++)
#pragma unroll
            for (int r = 0; r < 8; r++) acc[i][j][r] = b;
    }

    int ldrow = tid >> 1;              // 0..127
    int ldcol = (tid & 1) * 16;        // 0 or 16 halves

    const _Float16* gA = Ae + (size_t)(bm * BM + ldrow) * K + ldcol;
    const _Float16* gB = Be + (size_t)(bn * BN + ldrow) * K + ldcol;
    unsigned int la[2], lb[2];
    la[0] = (unsigned int)(uintptr_t)&As[0][ldrow * BK + ldcol];
    lb[0] = (unsigned int)(uintptr_t)&Bs[0][ldrow * BK + ldcol];
    la[1] = (unsigned int)(uintptr_t)&As[1][ldrow * BK + ldcol];
    lb[1] = (unsigned int)(uintptr_t)&Bs[1][ldrow * BK + ldcol];

    auto compute_tile = [&](int cur) {
        FragU af[2];
#pragma unroll
        for (int i = 0; i < 2; i++) {
            int r  = wm * 32 + i * 16 + l16;
            int cb = lhi * 8;
            af[i].u[0] = *(const uint4*)&As[cur][r * BK + cb];
            af[i].u[1] = *(const uint4*)&As[cur][r * BK + cb + 16];
        }
        FragU bf[4];
#pragma unroll
        for (int j = 0; j < 4; j++) {
            int r  = wn * 64 + j * 16 + l16;
            int cb = lhi * 8;
            bf[j].u[0] = *(const uint4*)&Bs[cur][r * BK + cb];
            bf[j].u[1] = *(const uint4*)&Bs[cur][r * BK + cb + 16];
        }
#pragma unroll
        for (int i = 0; i < 2; i++)
#pragma unroll
            for (int j = 0; j < 4; j++)
                acc[i][j] = __builtin_amdgcn_wmma_f32_16x16x32_f16(
                    false, af[i].h, false, bf[j].h, (short)0, acc[i][j],
                    false, false);
    };

    // prologue: stage k=0 into buffer 0
    async_stage32(la[0], gA);
    async_stage32(lb[0], gB);

    int nsteps = K / BK;
    // steady state: branch-free issue / wait<=4 / compute
    for (int s = 0; s < nsteps - 1; ++s) {
        int cur = s & 1;
        async_stage32(la[cur ^ 1], gA + (size_t)(s + 1) * BK);
        async_stage32(lb[cur ^ 1], gB + (size_t)(s + 1) * BK);
        asm volatile("s_wait_asynccnt 0x4" ::: "memory");
        __syncthreads();
        compute_tile(cur);
        __syncthreads();   // protect buffer before it is re-staged
    }
    // tail
    asm volatile("s_wait_asynccnt 0x0" ::: "memory");
    __syncthreads();
    compute_tile((nsteps - 1) & 1);

    int mbase = bm * BM + wm * 32;
    int nbase = bn * BN + wn * 64;
#pragma unroll
    for (int i = 0; i < 2; i++) {
#pragma unroll
        for (int j = 0; j < 4; j++) {
            int n = nbase + j * 16 + l16;
#pragma unroll
            for (int r = 0; r < 8; r++) {
                int m = mbase + i * 16 + lhi * 8 + r;
                float v = acc[i][j][r];
                if (MODE == 1) {
                    v = v > 0.0f ? v : 0.0f;
                    Hout[(size_t)e * M * N + (size_t)m * N + n] = (_Float16)v;
                } else {
                    float g = gates[e * CAP + m];
                    if (g != 0.0f) {
                        int t = tok[e * CAP + m];
                        Yout[(size_t)t * N + n] = v * g;   // rows unique (top-1)
                    }
                }
            }
        }
    }
}

// ---------------------------------------------------------------------------
// 6. Load-balancing loss
// ---------------------------------------------------------------------------
__global__ void loss_kernel(const int* __restrict__ list_cnt,
                            const float* __restrict__ probsum,
                            float* __restrict__ out_loss) {
    if (threadIdx.x == 0) {
        float s = 0.0f;
        for (int e = 0; e < NEXP; e++)
            s += ((float)list_cnt[e] / (float)T_TOK) *
                 (probsum[e] / (float)T_TOK);
        *out_loss = (float)NEXP * s;
    }
}

// ---------------------------------------------------------------------------
extern "C" void kernel_launch(void* const* d_in, const int* in_sizes, int n_in,
                              void* d_out, int out_size, void* d_ws, size_t ws_size,
                              hipStream_t stream) {
    (void)in_sizes; (void)n_in; (void)ws_size;
    const float* x  = (const float*)d_in[0];   // [T, D]
    const float* Wr = (const float*)d_in[1];   // [D, E]
    const float* W1 = (const float*)d_in[2];   // [E, D, F]
    const float* b1 = (const float*)d_in[3];   // [E, F]
    const float* W2 = (const float*)d_in[4];   // [E, F, D]
    const float* b2 = (const float*)d_in[5];   // [E, D]
    float* out = (float*)d_out;                // [T*D] ++ [loss]

    char* ws = (char*)d_ws;
    const size_t SZ_WT = (size_t)NEXP * D_MODEL * D_FF * sizeof(_Float16); // 64 MB
    _Float16* W1T = (_Float16*)ws;                 ws += SZ_WT;  // [E][F][D]
    _Float16* W2T = (_Float16*)ws;                 ws += SZ_WT;  // [E][D][F]
    _Float16* H   = (_Float16*)ws;                 ws += SZ_WT;  // [E][CAP][F]
    _Float16* Xe  = (_Float16*)ws;                 ws += (size_t)NEXP * CAP * D_MODEL * 2; // [E][CAP][D]
    unsigned long long* plist = (unsigned long long*)ws; ws += (size_t)NEXP * T_TOK * 8;
    float* gates  = (float*)ws;                    ws += (size_t)NEXP * CAP * 4;
    int*   tok    = (int*)ws;                      ws += (size_t)NEXP * CAP * 4;
    int*   list_cnt = (int*)ws;                    ws += NEXP * 4;
    float* probsum  = (float*)ws;

    hipMemsetAsync(d_out, 0, (size_t)out_size * sizeof(float), stream);
    hipMemsetAsync(list_cnt, 0, NEXP * 4 * 2, stream);  // counters + probsums

    // weight convert/transpose: W1 [D][F] -> [F][D], W2 [F][D] -> [D][F]
    transpose_cvt_kernel<<<dim3(D_FF / 32, D_MODEL / 32, NEXP), dim3(32, 8), 0, stream>>>(
        W1, W1T, D_MODEL, D_FF);
    transpose_cvt_kernel<<<dim3(D_MODEL / 32, D_FF / 32, NEXP), dim3(32, 8), 0, stream>>>(
        W2, W2T, D_FF, D_MODEL);

    router_kernel<<<T_TOK / 8, 256, 0, stream>>>(x, Wr, list_cnt, probsum, plist);
    loss_kernel<<<1, 32, 0, stream>>>(list_cnt, probsum, out + (size_t)T_TOK * D_MODEL);

    sort_kernel<<<NEXP, 1024, 0, stream>>>(plist, list_cnt, gates, tok);
    gather_kernel<<<NEXP * CAP, 256, 0, stream>>>(x, tok, Xe);

    // GEMM1: [CAP x D] x [D x F] -> H (relu, f16)
    moe_gemm_kernel<1><<<dim3(D_FF / BN, CAP / BM, NEXP), 256, 0, stream>>>(
        Xe, W1T, b1, gates, tok, H, nullptr, CAP, D_FF, D_MODEL);
    // GEMM2: [CAP x F] x [F x D] -> scatter to out (gated)
    moe_gemm_kernel<2><<<dim3(D_MODEL / BN, CAP / BM, NEXP), 256, 0, stream>>>(
        H, W2T, b2, gates, tok, nullptr, out, CAP, D_MODEL, D_FF);
}
